// MultiHeadAttention_15255723835433
// MI455X (gfx1250) — matrix operand order repaired and checked by reading, not compiled, not run on hardware
//
#include <hip/hip_runtime.h>

#define B_  4
#define H_  16
#define D_  64
#define LQ_ 2048
#define LK_ 2048
#define IN_ 1024

typedef __attribute__((ext_vector_type(8)))  __bf16 v8bf;
typedef __attribute__((ext_vector_type(16))) __bf16 v16bf;
typedef __attribute__((ext_vector_type(8)))  float  v8f;

__device__ __forceinline__ unsigned short f2bf(float f) {
  unsigned int u = __float_as_uint(f);
  u += 0x7FFFu + ((u >> 16) & 1u);   // round-to-nearest-even
  return (unsigned short)(u >> 16);
}

// Build a 16-element bf16 A-operand from two 8-element contiguous chunks.
__device__ __forceinline__ v16bf load_a16(const unsigned short* p0,
                                          const unsigned short* p1) {
  v8bf lo = *reinterpret_cast<const v8bf*>(p0);
  v8bf hi = *reinterpret_cast<const v8bf*>(p1);
  return __builtin_shufflevector(lo, hi, 0,1,2,3,4,5,6,7,8,9,10,11,12,13,14,15);
}

__device__ __forceinline__ v16bf load_b16(const unsigned short* p) {
  return *reinterpret_cast<const v16bf*>(p);   // 16 contiguous bf16 (32B)
}

#define WMMA_BF16(a, b, c) \
  __builtin_amdgcn_wmma_f32_16x16x32_bf16(false, (a), false, (b), (short)0, (c), false, false)

// ---------------------------------------------------------------- convert (vectorized)
__global__ void cvt_bf16_kernel(const float* __restrict__ src,
                                unsigned short* __restrict__ dst, int n4) {
  int i = blockIdx.x * blockDim.x + threadIdx.x;
  int stride = gridDim.x * blockDim.x;
  for (; i < n4; i += stride) {
    float4 f = reinterpret_cast<const float4*>(src)[i];
    ushort4 o;
    o.x = f2bf(f.x); o.y = f2bf(f.y); o.z = f2bf(f.z); o.w = f2bf(f.w);
    reinterpret_cast<ushort4*>(dst)[i] = o;
  }
}

// ---------------------------------------------------------------- mask bit-packing
// mask int32 [LQ, LK] -> bitmap uint32 [LQ, LK/32]
__global__ void pack_mask_kernel(const int* __restrict__ mask,
                                 unsigned int* __restrict__ mp) {
  int w = blockIdx.x * blockDim.x + threadIdx.x;        // 131072 words
  if (w >= LQ_ * (LK_ / 32)) return;
  const int* src = mask + (size_t)w * 32;
  unsigned int bits = 0;
  #pragma unroll
  for (int i = 0; i < 32; i += 4) {
    int4 m4 = *reinterpret_cast<const int4*>(src + i);
    bits |= (m4.x ? 1u : 0u) << (i + 0);
    bits |= (m4.y ? 1u : 0u) << (i + 1);
    bits |= (m4.z ? 1u : 0u) << (i + 2);
    bits |= (m4.w ? 1u : 0u) << (i + 3);
  }
  mp[w] = bits;
}

// ---------------------------------------------------------------- projection GEMM
// y[m,n] = sum_k x[m,k] * W[n,k] + bias[n], scaled; one wave = 16 rows x 64 cols (one head)
__global__ void __launch_bounds__(128)
proj_wmma_kernel(const unsigned short* __restrict__ xbf,   // [8192, 1024] bf16
                 const unsigned short* __restrict__ wbf,   // [1024, 1024] bf16
                 const float* __restrict__ bias,           // [1024]
                 unsigned short* __restrict__ outp,        // [B,H,L,D] or [B,H,D,L]
                 int transposed, float scale) {
  const int lane = threadIdx.x & 31;
  const int wave = threadIdx.x >> 5;
  const int hh   = lane >> 4;     // which 16-lane half
  const int nl   = lane & 15;
  const int m0   = blockIdx.x * 16;
  const int head = blockIdx.y * 4 + wave;
  const int n0   = head * 64;

  v8f acc[4];
  #pragma unroll
  for (int t = 0; t < 4; ++t) acc[t] = (v8f){};

  const unsigned short* arow = xbf + (size_t)(m0 + nl) * IN_;

  for (int kk = 0; kk < IN_; kk += 32) {
    // A: row = nl; lanes 0-15 hold K kk+0..15, lanes 16-31 hold kk+16..31
    v16bf a = load_a16(arow + kk + hh * 16,
                       arow + kk + hh * 16 + 8);
    #pragma unroll
    for (int t = 0; t < 4; ++t) {
      // B[k, n] = W[n0+16t+nl, k]; lane element i -> k = kk + hh*16 + i (contiguous)
      v16bf bb = load_b16(wbf + (size_t)(n0 + 16 * t + nl) * IN_ + kk + hh * 16);
      acc[t] = WMMA_BF16(a, bb, acc[t]);
    }
  }

  #pragma unroll
  for (int t = 0; t < 4; ++t) {
    #pragma unroll
    for (int j = 0; j < 8; ++j) {
      int m  = m0 + j + 8 * hh;            // global row
      int b  = m >> 11;
      int l  = m & 2047;
      int dl = t * 16 + nl;                // dim within head
      float v = (acc[t][j] + bias[n0 + dl]) * scale;
      size_t idx;
      if (!transposed) idx = ((size_t)(b * H_ + head) * LQ_ + l) * D_ + dl;
      else             idx = ((size_t)(b * H_ + head) * D_ + dl) * (size_t)LK_ + l;
      outp[idx] = f2bf(v);
    }
  }
}

// ---------------------------------------------------------------- attention core
// One block = (b, h, 32 q-rows). Waves 0-3 -> rows 0-15, waves 4-7 -> rows 16-31.
__global__ void __launch_bounds__(256)
attn_wmma_kernel(const unsigned short* __restrict__ qh,    // [B,H,LQ,D] bf16 (pre-scaled)
                 const unsigned short* __restrict__ kh,    // [B,H,LK,D] bf16
                 const unsigned short* __restrict__ vhT,   // [B,H,D,LK] bf16
                 const unsigned int* __restrict__ mbits,   // [LQ, 64] bit-packed mask
                 float* __restrict__ scores,               // [B,H,LQ,LK]
                 float* __restrict__ U) {                  // [B,H,LQ,D]
  const int lane = threadIdx.x & 31;
  const int wave = threadIdx.x >> 5;
  const int hh   = lane >> 4;
  const int nl   = lane & 15;
  const int qs   = wave >> 2;      // q-subtile (0/1)
  const int wk   = wave & 3;       // k-strip within subtile group

  const int blk = blockIdx.x;      // B*H*64
  const int qt  = blk & 63;
  const int h   = (blk >> 6) & 15;
  const int b   = blk >> 10;
  const int qb  = qt * 32;
  const int q0  = qb + qs * 16;
  const size_t bh = (size_t)(b * H_ + h);

  __shared__ float rowsum[32];
  __shared__ __align__(16) float Uacc[32][64];
  __shared__ unsigned int mrow[32][64];                 // block's mask bits
  __shared__ unsigned short pbuf[8][16][32];

  if (threadIdx.x < 32) rowsum[threadIdx.x] = 0.f;
  for (int i = threadIdx.x; i < 32 * 64; i += 256) ((float*)Uacc)[i] = 0.f;
  for (int i = threadIdx.x; i < 32 * 64; i += 256)
    ((unsigned int*)mrow)[i] = mbits[(size_t)(qb + (i >> 6)) * 64 + (i & 63)];
  __syncthreads();

  // Q A-operands held in registers for the whole kernel (d = 0..31 and 32..63)
  const unsigned short* qrow = qh + (bh * LQ_ + q0 + nl) * D_;
  v16bf aq0 = load_a16(qrow + hh * 16,      qrow + hh * 16 + 8);
  v16bf aq1 = load_a16(qrow + 32 + hh * 16, qrow + 40 + hh * 16);

  // ---- pass 1: per-row sum of exp (masked entries contribute 0) ----
  float rs[8];
  #pragma unroll
  for (int j = 0; j < 8; ++j) rs[j] = 0.f;

  for (int kt = wk; kt < LK_ / 16; kt += 4) {
    int k0 = kt * 16;
    const unsigned short* krow = kh + (bh * LK_ + k0 + nl) * D_;
    __builtin_prefetch(krow + 64 * D_, 0, 3);           // next strip (L2 -> near)
    v16bf bk0 = load_b16(krow + hh * 16);
    v16bf bk1 = load_b16(krow + 32 + hh * 16);
    v8f c = (v8f){};
    c = WMMA_BF16(aq0, bk0, c);
    c = WMMA_BF16(aq1, bk1, c);
    int col = k0 + nl;
    #pragma unroll
    for (int j = 0; j < 8; ++j) {
      int rr = qs * 16 + j + 8 * hh;
      unsigned int mv = (mrow[rr][col >> 5] >> (col & 31)) & 1u;
      rs[j] += mv ? __expf(c[j]) : 0.f;
    }
  }
  #pragma unroll
  for (int j = 0; j < 8; ++j) {
    #pragma unroll
    for (int off = 1; off < 16; off <<= 1) rs[j] += __shfl_xor(rs[j], off, 32);
  }
  if (nl == 0) {
    #pragma unroll
    for (int j = 0; j < 8; ++j) atomicAdd(&rowsum[qs * 16 + j + 8 * hh], rs[j]);
  }
  __syncthreads();

  float inv[8];
  #pragma unroll
  for (int j = 0; j < 8; ++j) {
    float s = rowsum[qs * 16 + j + 8 * hh];
    inv[j] = s > 0.f ? 1.f / s : 0.f;
  }

  // ---- pass 2: write normalized scores once, accumulate U = P @ V ----
  v8f cU[4];
  #pragma unroll
  for (int t = 0; t < 4; ++t) cU[t] = (v8f){};

  for (int pr = wk; pr < LK_ / 32; pr += 4) {
    int k0 = pr * 32;
    #pragma unroll
    for (int sub = 0; sub < 2; ++sub) {
      int ks = k0 + sub * 16;
      const unsigned short* krow = kh + (bh * LK_ + ks + nl) * D_;
      v16bf bk0 = load_b16(krow + hh * 16);
      v16bf bk1 = load_b16(krow + 32 + hh * 16);
      v8f c = (v8f){};
      c = WMMA_BF16(aq0, bk0, c);
      c = WMMA_BF16(aq1, bk1, c);
      int col = ks + nl;
      #pragma unroll
      for (int j = 0; j < 8; ++j) {
        int rl  = j + 8 * hh;
        int rr  = qs * 16 + rl;
        unsigned int mv = (mrow[rr][col >> 5] >> (col & 31)) & 1u;
        float p = mv ? __expf(c[j]) * inv[j] : 0.f;
        scores[(bh * LQ_ + q0 + rl) * LK_ + col] = p;    // final prob, written once
        pbuf[wave][rl][sub * 16 + nl] = f2bf(p);         // repack for A layout
      }
    }
    asm volatile("s_wait_dscnt 0" ::: "memory");         // LDS store -> load, same wave
    v16bf ap = load_a16(&pbuf[wave][nl][hh * 16],
                        &pbuf[wave][nl][hh * 16 + 8]);
    #pragma unroll
    for (int t = 0; t < 4; ++t) {
      // B[kk, d] = V[k0+kk, t*16+nl] = vhT[(bh*64 + t*16+nl)*LK + k0 + kk] contiguous
      const unsigned short* vrow = vhT + (bh * D_ + t * 16 + nl) * (size_t)LK_ + k0 + hh * 16;
      __builtin_prefetch(vrow + 128, 0, 3);              // next pair for this wave
      v16bf bv = load_b16(vrow);
      cU[t] = WMMA_BF16(ap, bv, cU[t]);
    }
  }

  #pragma unroll
  for (int t = 0; t < 4; ++t) {
    #pragma unroll
    for (int j = 0; j < 8; ++j)
      atomicAdd(&Uacc[qs * 16 + j + 8 * hh][t * 16 + nl], cU[t][j]);
  }
  __syncthreads();

  for (int i = threadIdx.x; i < 512; i += 256) {         // 32 rows x 64 cols, float4
    int r  = i >> 4;
    int d4 = (i & 15) * 4;
    float4 uv = *reinterpret_cast<float4*>(&Uacc[r][d4]);
    *reinterpret_cast<float4*>(&U[(bh * LQ_ + qb + r) * D_ + d4]) = uv;
  }
}

// ---------------------------------------------------------------- output projection
__global__ void __launch_bounds__(64)
outproj_kernel(const float* __restrict__ U,    // [B,H,LQ,D]
               const float* __restrict__ wo,   // [64, 1024]
               const float* __restrict__ bo,   // [64]
               float* __restrict__ out) {      // [B,LQ,64]
  int row = blockIdx.x;            // b*2048 + lq
  int b = row >> 11, lq = row & 2047;
  __shared__ float f[1024];
  int t = threadIdx.x;
  for (int h = 0; h < 16; ++h)
    f[h * 64 + t] = U[(((size_t)(b * 16 + h)) * LQ_ + lq) * D_ + t];
  __syncthreads();
  float acc = bo[t];
  const float* wrow = wo + (size_t)t * 1024;
  #pragma unroll 4
  for (int i = 0; i < 1024; i += 4) {
    float4 wv = *reinterpret_cast<const float4*>(wrow + i);
    float4 fv = *reinterpret_cast<const float4*>(&f[i]);
    acc += fv.x * wv.x + fv.y * wv.y + fv.z * wv.z + fv.w * wv.w;
  }
  out[(size_t)row * 64 + t] = acc;
}

// ---------------------------------------------------------------- launch
extern "C" void kernel_launch(void* const* d_in, const int* in_sizes, int n_in,
                              void* d_out, int out_size, void* d_ws, size_t ws_size,
                              hipStream_t stream) {
  const float* q    = (const float*)d_in[0];
  const float* k    = (const float*)d_in[1];
  const float* v    = (const float*)d_in[2];
  const int*   mask = (const int*)  d_in[3];
  const float* wq   = (const float*)d_in[4];
  const float* bq   = (const float*)d_in[5];
  const float* wk   = (const float*)d_in[6];
  const float* bk   = (const float*)d_in[7];
  const float* wv   = (const float*)d_in[8];
  const float* bv   = (const float*)d_in[9];
  const float* wo   = (const float*)d_in[10];
  const float* bo   = (const float*)d_in[11];

  const size_t MXI = (size_t)B_ * LQ_ * IN_;     // 8,388,608
  const size_t WSZ = (size_t)IN_ * IN_;          // 1,048,576
  const size_t HSZ = (size_t)B_ * H_ * LQ_ * D_; // 8,388,608

  unsigned short* qbf  = (unsigned short*)d_ws;
  unsigned short* kbf  = qbf  + MXI;
  unsigned short* vbf  = kbf  + MXI;
  unsigned short* wqbf = vbf  + MXI;
  unsigned short* wkbf = wqbf + WSZ;
  unsigned short* wvbf = wkbf + WSZ;
  unsigned short* qhb  = wvbf + WSZ;
  unsigned short* khb  = qhb  + HSZ;
  unsigned short* vtb  = khb  + HSZ;
  float*          U    = (float*)(vtb + HSZ);    // 16B-aligned byte offset
  unsigned int*   mp   = (unsigned int*)(U + HSZ);

  cvt_bf16_kernel<<<1024, 256, 0, stream>>>(q,  qbf,  (int)(MXI / 4));
  cvt_bf16_kernel<<<1024, 256, 0, stream>>>(k,  kbf,  (int)(MXI / 4));
  cvt_bf16_kernel<<<1024, 256, 0, stream>>>(v,  vbf,  (int)(MXI / 4));
  cvt_bf16_kernel<<<256,  256, 0, stream>>>(wq, wqbf, (int)(WSZ / 4));
  cvt_bf16_kernel<<<256,  256, 0, stream>>>(wk, wkbf, (int)(WSZ / 4));
  cvt_bf16_kernel<<<256,  256, 0, stream>>>(wv, wvbf, (int)(WSZ / 4));
  pack_mask_kernel<<<512, 256, 0, stream>>>(mask, mp);

  dim3 pg(512, 4);  // 512 row-tiles x (4 blocks * 4 waves = 16 heads)
  proj_wmma_kernel<<<pg, 128, 0, stream>>>(qbf, wqbf, bq, qhb, 0, 0.125f);
  proj_wmma_kernel<<<pg, 128, 0, stream>>>(kbf, wkbf, bk, khb, 0, 1.0f);
  proj_wmma_kernel<<<pg, 128, 0, stream>>>(vbf, wvbf, bv, vtb, 1, 1.0f);

  float* out    = (float*)d_out;
  float* scores = out + (size_t)B_ * LQ_ * D_;
  attn_wmma_kernel<<<B_ * H_ * (LQ_ / 32), 256, 0, stream>>>(qhb, khb, vtb, mp, scores, U);
  outproj_kernel<<<B_ * LQ_, 64, 0, stream>>>(U, wo, bo, out);
}
